// Model_53953379172367
// MI455X (gfx1250) — matrix operand order repaired and checked by reading, not compiled, not run on hardware
//
#include <hip/hip_runtime.h>
#include <hip/hip_bf16.h>
#include <math.h>

// ---------------------------------------------------------------------------
// Model dims (compile-time constants from the reference)
// ---------------------------------------------------------------------------
#define BATCH   8
#define SEQL    1024
#define ROWS    (BATCH * SEQL)   // 8192
#define ENCIN   48
#define DMODEL  256
#define DINNER  512
#define DSTATE  16
#define DTRANK  16

typedef _Float16 half_t;
typedef _Float16 v16h  __attribute__((ext_vector_type(16)));
typedef _Float16 half8 __attribute__((ext_vector_type(8)));
typedef float    v8f   __attribute__((ext_vector_type(8)));
typedef float    f4v   __attribute__((ext_vector_type(4)));

// ---------------------------------------------------------------------------
// Split-precision WMMA GEMM:  C[M,N] = act(A[M,K] @ W[N,K]^T + bias)
// f32 operands split on the fly into f16 hi/lo; 3x v_wmma_f32_16x16x32_f16
// per K-tile per output subtile (Ah*Bh + Ah*Bl + Al*Bh) ~= f32 accuracy on
// the fast f16 matrix path.
//
// Block tile 64(M) x 64(N) x 32(K); 8 waves; each wave owns a 16x32 strip
// (2 accumulators, 6 WMMAs/K-step, A-fragment reused across both subtiles).
// Staging is branch-free: aligned runs of 8 floats (2x b128 loads), run
// either fully in- or out-of-range (requires K % 8 == 0, M % 64 == 0 --
// true for every GEMM in this model). B is stored transposed [n][k] in LDS
// so all fragment loads are contiguous 16B ds_load_b128.
// act: 0 = none, 1 = exact GELU (erf), 2 = softplus
// ---------------------------------------------------------------------------
#define BM 64
#define BN 64
#define BK 32
#define LPAD 48   // LDS row stride in halves (96 B = 16B multiple)

__global__ __launch_bounds__(256) void gemm_wmma_kernel(
    const float* __restrict__ A, int lda,
    const float* __restrict__ W,      // [N,K] row-major
    const float* __restrict__ bias,   // [N] or nullptr
    float* __restrict__ C, int ldc,
    int M, int N, int K, int act)
{
    __shared__ half_t Ah[BM][LPAD], Al[BM][LPAD];   // [m][k]
    __shared__ half_t Bh[BN][LPAD], Bl[BN][LPAD];   // [n][k] (transposed)

    const int tid  = threadIdx.x;
    const int lane = tid & 31;
    const int wave = tid >> 5;
    const int wr   = wave & 3;          // 0..3 : 16-row group
    const int wc   = wave >> 2;         // 0..1 : 32-col group
    const int m0   = blockIdx.x * BM;
    const int n0   = blockIdx.y * BN;
    const int half_sel = lane >> 4;
    const int lanelo   = lane & 15;

    const int sr = tid >> 2;            // staging row   0..63
    const int sc = (tid & 3) * 8;       // staging col   0,8,16,24

    v8f acc0 = {}, acc1 = {};
    const f4v zero4 = {};

    for (int k0 = 0; k0 < K; k0 += BK) {
        // ---- stage A (64x32): branch-free 8-float run ----
        {
            bool ok = (k0 + sc) < K;                         // run all-in or all-out
            size_t off = ok ? ((size_t)(m0 + sr) * lda + (size_t)(k0 + sc)) : 0;
            f4v v0 = *(const f4v*)(A + off);
            f4v v1 = *(const f4v*)(A + off + 4);
            v0 = ok ? v0 : zero4;
            v1 = ok ? v1 : zero4;
            half8 h, l;
            #pragma unroll
            for (int j = 0; j < 8; ++j) {
                float f = (j < 4) ? v0[j] : v1[j - 4];
                half_t hh = (half_t)f;
                h[j] = hh;
                l[j] = (half_t)(f - (float)hh);
            }
            *(half8*)&Ah[sr][sc] = h;
            *(half8*)&Al[sr][sc] = l;
        }
        // ---- stage B transposed (64n x 32k): B[n][k] = W[n0+n][k0+k] ----
        {
            bool ok = ((k0 + sc) < K) && ((n0 + sr) < N);
            size_t off = ok ? ((size_t)(n0 + sr) * K + (size_t)(k0 + sc)) : 0;
            f4v v0 = *(const f4v*)(W + off);
            f4v v1 = *(const f4v*)(W + off + 4);
            v0 = ok ? v0 : zero4;
            v1 = ok ? v1 : zero4;
            half8 h, l;
            #pragma unroll
            for (int j = 0; j < 8; ++j) {
                float f = (j < 4) ? v0[j] : v1[j - 4];
                half_t hh = (half_t)f;
                h[j] = hh;
                l[j] = (half_t)(f - (float)hh);
            }
            *(half8*)&Bh[sr][sc] = h;
            *(half8*)&Bl[sr][sc] = l;
        }
        __syncthreads();

        // ---- A fragment: lane holds row m, K runs [h*8..+7] and [16+h*8..+7]
        const int am = wr * 16 + lanelo;
        half8 a0h = *(const half8*)&Ah[am][half_sel * 8];
        half8 a1h = *(const half8*)&Ah[am][16 + half_sel * 8];
        half8 a0l = *(const half8*)&Al[am][half_sel * 8];
        half8 a1l = *(const half8*)&Al[am][16 + half_sel * 8];
        v16h a_h = __builtin_shufflevector(a0h, a1h,
            0,1,2,3,4,5,6,7,8,9,10,11,12,13,14,15);
        v16h a_l = __builtin_shufflevector(a0l, a1l,
            0,1,2,3,4,5,6,7,8,9,10,11,12,13,14,15);

        // ---- two 16-wide N subtiles per wave (A reused) ----
        #pragma unroll
        for (int nt = 0; nt < 2; ++nt) {
            const int bn = wc * 32 + nt * 16 + lanelo;   // lane's column
            // B fragment: K runs [h*8..+7] and [16+h*8..+7], matching A
            half8 b0h = *(const half8*)&Bh[bn][half_sel * 8];
            half8 b1h = *(const half8*)&Bh[bn][16 + half_sel * 8];
            half8 b0l = *(const half8*)&Bl[bn][half_sel * 8];
            half8 b1l = *(const half8*)&Bl[bn][16 + half_sel * 8];
            v16h b_h = __builtin_shufflevector(b0h, b1h,
                0,1,2,3,4,5,6,7,8,9,10,11,12,13,14,15);
            v16h b_l = __builtin_shufflevector(b0l, b1l,
                0,1,2,3,4,5,6,7,8,9,10,11,12,13,14,15);

            v8f acc = nt ? acc1 : acc0;
            acc = __builtin_amdgcn_wmma_f32_16x16x32_f16(false, a_h, false, b_h,
                                                         (short)0, acc, false, false);
            acc = __builtin_amdgcn_wmma_f32_16x16x32_f16(false, a_h, false, b_l,
                                                         (short)0, acc, false, false);
            acc = __builtin_amdgcn_wmma_f32_16x16x32_f16(false, a_l, false, b_h,
                                                         (short)0, acc, false, false);
            if (nt) acc1 = acc; else acc0 = acc;
        }
        __syncthreads();
    }

    // ---- epilogue: C/D layout M = q + 8*(lane/16), N = lane%16 ----
    #pragma unroll
    for (int nt = 0; nt < 2; ++nt) {
        int n = n0 + wc * 32 + nt * 16 + lanelo;
        if (n < N) {                                   // M % 64 == 0: no m guard
            v8f acc = nt ? acc1 : acc0;
            float bv = bias ? bias[n] : 0.f;
            #pragma unroll
            for (int q = 0; q < 8; ++q) {
                int m = m0 + wr * 16 + q + (half_sel << 3);
                float v = acc[q] + bv;
                if (act == 1) {                        // exact GELU
                    v = 0.5f * v * (1.f + erff(v * 0.70710678118654752f));
                } else if (act == 2) {                 // softplus
                    v = (v > 20.f) ? v : log1pf(__expf(v));
                }
                C[m * ldc + n] = v;
            }
        }
    }
}

// ---------------------------------------------------------------------------
// Embed fixup: mask-token substitution + sinusoidal positional encoding
// ---------------------------------------------------------------------------
__global__ __launch_bounds__(256) void embed_fix_kernel(
    float* __restrict__ X, const unsigned char* __restrict__ mask,
    const float* __restrict__ mtok)
{
    int idx = blockIdx.x * 256 + threadIdx.x;
    if (idx >= ROWS * DMODEL) return;
    int row = idx >> 8, c = idx & 255;
    int l = row % SEQL;
    float v = X[idx];
    if (mask[row]) v = mtok[c];
    // div[j] = exp(-2j * ln(10000)/256); ang = l * div
    float ang = (float)l * __expf(-(float)(c & ~1) * 0.03597789207803197f);
    v += (c & 1) ? cosf(ang) : sinf(ang);
    X[idx] = v;
}

// ---------------------------------------------------------------------------
// Causal depthwise conv (4-tap, left pad 3) + SiLU.  xz: [ROWS, 2*DINNER]
// ---------------------------------------------------------------------------
__global__ __launch_bounds__(256) void conv_silu_kernel(
    const float* __restrict__ xz, const float* __restrict__ cw,
    const float* __restrict__ cb, float* __restrict__ out)
{
    int idx = blockIdx.x * 256 + threadIdx.x;
    if (idx >= ROWS * DINNER) return;
    int bl = idx >> 9, d = idx & 511;
    int b = bl / SEQL, l = bl % SEQL;
    float acc = cb[d];
    #pragma unroll
    for (int i = 0; i < 4; ++i) {
        int ll = l - 3 + i;
        float x = xz[(b * SEQL + max(ll, 0)) * (2 * DINNER) + d];
        acc += (ll >= 0) ? cw[d * 4 + i] * x : 0.f;
    }
    out[idx] = acc / (1.f + __expf(-acc));   // silu
}

// ---------------------------------------------------------------------------
// Selective scan. One thread per (b, d) channel; 16 SSM states in registers
// (16 independent v_exp_f32 + FMA chains per step hide pipeline latency).
// y aliases dt: each thread reads-then-writes only its own element.
// ---------------------------------------------------------------------------
__global__ __launch_bounds__(256) void scan_kernel(
    const float* __restrict__ u,   // [ROWS, DINNER]
    const float* dt,               // [ROWS, DINNER]   (aliases y)
    const float* __restrict__ bc,  // [ROWS, 48] : [dt_rank | B | C]
    const float* __restrict__ A_log, const float* __restrict__ Dp,
    float* y)
{
    int idx = blockIdx.x * 256 + threadIdx.x;
    if (idx >= BATCH * DINNER) return;
    int b = idx / DINNER, d = idx % DINNER;

    float a[DSTATE], h[DSTATE];
    #pragma unroll
    for (int n = 0; n < DSTATE; ++n) {
        a[n] = -__expf(A_log[d * DSTATE + n]);
        h[n] = 0.f;
    }
    float Dd = Dp[d];
    int base = b * SEQL;
    for (int l = 0; l < SEQL; ++l) {
        int row = base + l;
        float dtv = dt[row * DINNER + d];
        float uv  = u[row * DINNER + d];
        float du  = dtv * uv;
        const float* bcr = bc + row * 48;
        float yv = 0.f;
        #pragma unroll
        for (int n = 0; n < DSTATE; ++n) {
            float dA = __expf(dtv * a[n]);
            h[n] = dA * h[n] + du * bcr[DTRANK + n];
            yv += h[n] * bcr[DTRANK + DSTATE + n];
        }
        y[row * DINNER + d] = yv + uv * Dd;
    }
}

// ---------------------------------------------------------------------------
// Gate: y *= silu(z), z = xz[:, 512:1024]
// ---------------------------------------------------------------------------
__global__ __launch_bounds__(256) void gate_kernel(
    float* __restrict__ y, const float* __restrict__ xz)
{
    int idx = blockIdx.x * 256 + threadIdx.x;
    if (idx >= ROWS * DINNER) return;
    int bl = idx >> 9, d = idx & 511;
    float z = xz[bl * (2 * DINNER) + DINNER + d];
    y[idx] *= z / (1.f + __expf(-z));
}

// ---------------------------------------------------------------------------
// Flip along sequence dim: out[b,l,:] = in[b,L-1-l,:]
// ---------------------------------------------------------------------------
__global__ __launch_bounds__(256) void flip_kernel(
    const float* __restrict__ in, float* __restrict__ out)
{
    int idx = blockIdx.x * 256 + threadIdx.x;
    if (idx >= ROWS * DMODEL) return;
    int row = idx >> 8, c = idx & 255;
    int b = row / SEQL, l = row % SEQL;
    out[idx] = in[(b * SEQL + (SEQL - 1 - l)) * DMODEL + c];
}

// ---------------------------------------------------------------------------
// Fused (x + add1 + maybe-flipped add2) -> LayerNorm(gamma, beta).
// One block per row (256 threads, D_MODEL=256). In-place safe (out == x).
// ---------------------------------------------------------------------------
__global__ __launch_bounds__(256) void add_ln_kernel(
    const float* __restrict__ x, const float* add1, const float* add2,
    int flip2, const float* __restrict__ g, const float* __restrict__ be,
    float* out)
{
    __shared__ float red[256];
    int row = blockIdx.x;
    int c = threadIdx.x;
    float v = x[row * DMODEL + c];
    if (add1) v += add1[row * DMODEL + c];
    if (add2) {
        int r2 = row;
        if (flip2) { int b = row / SEQL, l = row % SEQL; r2 = b * SEQL + (SEQL - 1 - l); }
        v += add2[r2 * DMODEL + c];
    }
    red[c] = v; __syncthreads();
    for (int s = 128; s > 0; s >>= 1) { if (c < s) red[c] += red[c + s]; __syncthreads(); }
    float mu = red[0] * (1.f / DMODEL);
    __syncthreads();
    float dv = v - mu;
    red[c] = dv * dv; __syncthreads();
    for (int s = 128; s > 0; s >>= 1) { if (c < s) red[c] += red[c + s]; __syncthreads(); }
    float var = red[0] * (1.f / DMODEL);
    out[row * DMODEL + c] = dv * rsqrtf(var + 1e-5f) * g[c] + be[c];
}

// ---------------------------------------------------------------------------
// Deterministic masked-MSE loss: block partials then single-block finalize.
// ---------------------------------------------------------------------------
__global__ __launch_bounds__(256) void loss_partial_kernel(
    const float* __restrict__ pred, const float* __restrict__ xenc,
    const unsigned char* __restrict__ mask,
    float* __restrict__ psse, float* __restrict__ pcnt)
{
    __shared__ float s1[256], s2[256];
    int tid = threadIdx.x;
    float sse = 0.f, cnt = 0.f;
    for (int row = blockIdx.x * 256 + tid; row < ROWS; row += gridDim.x * 256) {
        if (mask[row]) {
            cnt += 1.f;
            const float* p = pred + row * ENCIN;
            const float* x = xenc + row * ENCIN;
            #pragma unroll 4
            for (int j = 0; j < ENCIN; ++j) { float d = p[j] - x[j]; sse += d * d; }
        }
    }
    s1[tid] = sse; s2[tid] = cnt; __syncthreads();
    for (int s = 128; s > 0; s >>= 1) {
        if (tid < s) { s1[tid] += s1[tid + s]; s2[tid] += s2[tid + s]; }
        __syncthreads();
    }
    if (tid == 0) { psse[blockIdx.x] = s1[0]; pcnt[blockIdx.x] = s2[0]; }
}

__global__ void loss_final_kernel(const float* psse, const float* pcnt,
                                  int nb, float* out)
{
    float sse = 0.f, cnt = 0.f;
    for (int i = 0; i < nb; ++i) { sse += psse[i]; cnt += pcnt[i]; }
    out[0] = sse / (cnt * (float)ENCIN);
}

// ---------------------------------------------------------------------------
// Host orchestration
// ---------------------------------------------------------------------------
struct MP {
    const float *in_w, *conv_w, *conv_b, *xproj_w, *dt_w, *dt_b, *A_log, *D, *out_w;
};

extern "C" void kernel_launch(void* const* d_in, const int* in_sizes, int n_in,
                              void* d_out, int out_size, void* d_ws, size_t ws_size,
                              hipStream_t stream)
{
    (void)in_sizes; (void)n_in; (void)out_size; (void)ws_size;

    // ---- input pytree, insertion-order flattened ----
    // 0:x_enc 1:mask 2:inp_w 3:inp_b 4:mask_token
    // per layer (28): mf[9], mb[9], n1_g,n1_b,fln_g,fln_b,f1_w,f1_b,f2_w,f2_b,n2_g,n2_b
    // 117:enc_g 118:enc_b 119..128: dec w1,c1,g1,gb1,w2,c2,g2,gb2,w3,c3
    auto PF = [&](int i) { return (const float*)d_in[i]; };
    const float* x_enc = PF(0);
    const unsigned char* mask = (const unsigned char*)d_in[1];   // bool, 1 byte
    const float* inp_w = PF(2);
    const float* inp_b = PF(3);
    const float* mtok  = PF(4);

    auto getMP = [&](int i) {
        MP p;
        p.in_w = PF(i); p.conv_w = PF(i + 1); p.conv_b = PF(i + 2);
        p.xproj_w = PF(i + 3); p.dt_w = PF(i + 4); p.dt_b = PF(i + 5);
        p.A_log = PF(i + 6); p.D = PF(i + 7); p.out_w = PF(i + 8);
        return p;
    };

    // ---- workspace layout (~103 MB of f32) ----
    float* X  = (float*)d_ws;
    float* XF = X  + ROWS * DMODEL;
    float* T0 = XF + ROWS * DMODEL;                 // [ROWS, 1024] xz
    float* T1 = T0 + ROWS * 2 * DINNER;             // [ROWS, 512]  xu (conv+silu)
    float* T2 = T1 + ROWS * DINNER;                 // [ROWS, 512]  dt -> y (in place)
    float* T3 = T2 + ROWS * DINNER;                 // [ROWS, 48]   dbc
    float* Y1 = T3 + ROWS * 48;                     // [ROWS, 256]
    float* Y2 = Y1 + ROWS * DMODEL;                 // [ROWS, 256]
    float* PS = Y2 + ROWS * DMODEL;                 // 32 partial sse
    float* PC = PS + 32;                            // 32 partial cnt

    auto gemm = [&](const float* A, int lda, const float* W, const float* bias,
                    float* C, int ldc, int M, int N, int K, int act) {
        dim3 grid(M / BM, (N + BN - 1) / BN);       // M % 64 == 0, K % 8 == 0
        gemm_wmma_kernel<<<grid, 256, 0, stream>>>(A, lda, W, bias, C, ldc, M, N, K, act);
    };
    const int EW = (ROWS * DINNER + 255) / 256;     // elementwise grid [ROWS,512]
    const int EM = (ROWS * DMODEL + 255) / 256;     // elementwise grid [ROWS,256]

    auto run_mamba = [&](const float* xin, const MP& p, float* yout) {
        // xz = x @ in_w^T                       [ROWS,1024]
        gemm(xin, DMODEL, p.in_w, nullptr, T0, 2 * DINNER, ROWS, 2 * DINNER, DMODEL, 0);
        // xu = silu(causal depthwise conv(xz[:, :512]))
        conv_silu_kernel<<<EW, 256, 0, stream>>>(T0, p.conv_w, p.conv_b, T1);
        // dbc = xu @ xproj_w^T                  [ROWS,48]
        gemm(T1, DINNER, p.xproj_w, nullptr, T3, 48, ROWS, 48, DINNER, 0);
        // dt = softplus(dbc[:, :16] @ dt_w^T + dt_b)   [ROWS,512]
        gemm(T3, 48, p.dt_w, p.dt_b, T2, DINNER, ROWS, DINNER, DTRANK, 2);
        // selective scan (y overwrites dt in place)
        scan_kernel<<<(BATCH * DINNER + 255) / 256, 256, 0, stream>>>(
            T1, T2, T3, p.A_log, p.D, T2);
        // y *= silu(z)
        gate_kernel<<<EW, 256, 0, stream>>>(T2, T0);
        // out = y @ out_w^T                     [ROWS,256]
        gemm(T2, DINNER, p.out_w, nullptr, yout, DMODEL, ROWS, DMODEL, DINNER, 0);
    };

    // ---- embedding: X = x_enc @ inp_w^T + inp_b; mask token; + pos enc ----
    gemm(x_enc, ENCIN, inp_w, inp_b, X, DMODEL, ROWS, DMODEL, ENCIN, 0);
    embed_fix_kernel<<<EM, 256, 0, stream>>>(X, mask, mtok);

    // ---- 4 bidirectional Mamba layers ----
    for (int layer = 0; layer < 4; ++layer) {
        int base = 5 + 28 * layer;
        MP mf = getMP(base);
        MP mb = getMP(base + 9);
        const float *n1_g = PF(base + 18), *n1_b = PF(base + 19);
        const float *fln_g = PF(base + 20), *fln_b = PF(base + 21);
        const float *f1_w = PF(base + 22), *f1_b = PF(base + 23);
        const float *f2_w = PF(base + 24), *f2_b = PF(base + 25);
        const float *n2_g = PF(base + 26), *n2_b = PF(base + 27);

        flip_kernel<<<EM, 256, 0, stream>>>(X, XF);
        run_mamba(X,  mf, Y1);                       // fwd
        run_mamba(XF, mb, Y2);                       // bwd (on flipped x)
        // x = LN(x + fwd + flip(bwd))
        add_ln_kernel<<<ROWS, 256, 0, stream>>>(X, Y1, Y2, 1, n1_g, n1_b, X);
        // h = LN(x); h = gelu(h@f1^T + b1)@f2^T + b2; x = LN(x + h)
        add_ln_kernel<<<ROWS, 256, 0, stream>>>(X, nullptr, nullptr, 0, fln_g, fln_b, Y1);
        gemm(Y1, DMODEL, f1_w, f1_b, T1, DINNER, ROWS, DINNER, DMODEL, 1);
        gemm(T1, DINNER, f2_w, f2_b, Y2, DMODEL, ROWS, DMODEL, DINNER, 0);
        add_ln_kernel<<<ROWS, 256, 0, stream>>>(X, Y2, nullptr, 0, n2_g, n2_b, X);
    }

    // ---- encoder norm + decoder ----
    const float *enc_g = PF(117), *enc_b = PF(118);
    const float *dw1 = PF(119), *dc1 = PF(120), *dg1 = PF(121), *dgb1 = PF(122);
    const float *dw2 = PF(123), *dc2 = PF(124), *dg2 = PF(125), *dgb2 = PF(126);
    const float *dw3 = PF(127), *dc3 = PF(128);

    add_ln_kernel<<<ROWS, 256, 0, stream>>>(X, nullptr, nullptr, 0, enc_g, enc_b, X);
    gemm(X, DMODEL, dw1, dc1, Y1, DMODEL, ROWS, DMODEL, DMODEL, 1);     // gelu
    add_ln_kernel<<<ROWS, 256, 0, stream>>>(Y1, nullptr, nullptr, 0, dg1, dgb1, Y1);
    gemm(Y1, DMODEL, dw2, dc2, Y2, DMODEL, ROWS, DMODEL, DMODEL, 1);    // gelu
    add_ln_kernel<<<ROWS, 256, 0, stream>>>(Y2, nullptr, nullptr, 0, dg2, dgb2, Y2);

    float* pred = (float*)d_out + 1;                // d_out = [loss | pred]
    gemm(Y2, DMODEL, dw3, dc3, pred, ENCIN, ROWS, ENCIN, DMODEL, 0);

    // ---- masked MSE loss (deterministic two-stage reduction) ----
    loss_partial_kernel<<<32, 256, 0, stream>>>(pred, x_enc, mask, PS, PC);
    loss_final_kernel<<<1, 1, 0, stream>>>(PS, PC, 32, (float*)d_out);
}
